// BarrierNet_47974784697159
// MI455X (gfx1250) — matrix operand order repaired and checked by reading, not compiled, not run on hardware
//
#include <hip/hip_runtime.h>
#include <math.h>

typedef __attribute__((ext_vector_type(16))) _Float16 v16h;
typedef __attribute__((ext_vector_type(8)))  float    v8f;

#define NW 8                    // waves per block (wave32)
#define THREADS (NW * 32)
#define ROWS_PER_BLOCK (NW * 16)

__global__ __launch_bounds__(THREADS)
void barriernet_wmma_kernel(
    const float* __restrict__ x,   const float* __restrict__ mean, const float* __restrict__ stdv,
    const float* __restrict__ W1,  const float* __restrict__ b1,
    const float* __restrict__ W21, const float* __restrict__ b21,
    const float* __restrict__ W22, const float* __restrict__ b22,
    const float* __restrict__ W31, const float* __restrict__ b31,
    const float* __restrict__ W32, const float* __restrict__ b32,
    const float* __restrict__ obstacles,
    float* __restrict__ out, int nB)
{
    // LDS: 4KB (W1) + 8KB + 8KB (f16 W21/W22) + 2.5KB (x) + 16.5KB*2 (scratch) ~= 55.5KB
    __shared__ float sW1[128][8];                                   // [k][0..4]=W1 row, [5]=b1[k]
    __shared__ __attribute__((aligned(32))) _Float16 sWh21[4096];   // [q][half][n][v][c] half pairs
    __shared__ __attribute__((aligned(32))) _Float16 sWh22[4096];
    __shared__ float sX[NW][16][5];                                 // raw x tile
    __shared__ float sS21[NW][16][33];                              // post-relu x21 tile (padded)
    __shared__ float sS22[NW][16][33];                              // post-relu x22 tile (padded)

    const int tid  = threadIdx.x;
    const int w    = tid >> 5;
    const int lane = tid & 31;
    const int half = lane >> 4;          // 0: lanes 0-15, 1: lanes 16-31
    const int r    = lane & 15;
    const long row0 = (long)blockIdx.x * ROWS_PER_BLOCK;

    // ---- stage layer-2 weights into LDS as f16 in WMMA B-fragment order.
    // B 32x16 f16 order used here (contiguous k, same as A): lanes 0-15 hold K=0..15 (VGPR v: K=2v,2v+1), lanes 16-31 K=16..31.
    for (int idx = tid; idx < 32 * 128; idx += THREADS) {
        int n = idx >> 7, k = idx & 127;                // idx == n*128 + k (row-major W[n][k])
        int q = k >> 5, kin = k & 31;                   // K chunk of 32
        int hb = kin >> 4, v = (kin & 15) >> 1, c = kin & 1;
        int dst = (((q * 2 + hb) * 32 + n) * 8 + v) * 2 + c;
        sWh21[dst] = (_Float16)W21[idx];
        sWh22[dst] = (_Float16)W22[idx];
    }
    // ---- stage W1 rows + b1
    if (tid < 128) {
        int k = tid;
        #pragma unroll
        for (int f = 0; f < 5; ++f) sW1[k][f] = W1[k * 5 + f];
        sW1[k][5] = b1[k];
    }
    // ---- stage raw x tile (h1 uses RAW x per the reference)
    for (int idx = tid; idx < ROWS_PER_BLOCK * 5; idx += THREADS) {
        int rr = idx / 5, f = idx - rr * 5;
        float v = (row0 + rr < nB) ? x[(row0 + rr) * 5 + f] : 0.0f;
        sX[rr >> 4][rr & 15][f] = v;
    }
    __syncthreads();

    // ---- layer 1 on VALU (exact f32), packed directly into f16 A-fragments.
    // A 16x32 f16 order used here (contiguous k, same as B): lane (half,r), VGPR v holds K = 32q + 16*half + 2*v (+0/+1).
    float xr[5];
    #pragma unroll
    for (int f = 0; f < 5; ++f) xr[f] = sX[w][r][f];

    v16h afr[4];
    #pragma unroll
    for (int q = 0; q < 4; ++q) {
        #pragma unroll
        for (int v = 0; v < 8; ++v) {
            const int klo = 32 * q + 16 * half + 2 * v;
            #pragma unroll
            for (int c = 0; c < 2; ++c) {
                const float* wr = sW1[klo + c];          // LDS broadcast (k uniform per half-wave)
                float a = wr[5];
                #pragma unroll
                for (int f = 0; f < 5; ++f) a = fmaf(xr[f], wr[f], a);
                afr[q][2 * v + c] = (_Float16)fmaxf(a, 0.0f);   // relu + cvt
            }
        }
    }

    // ---- dominant GEMMs on matrix pipe: 16 x v_wmma_f32_16x16x32_f16 per wave tile
    v8f a21_0 = {}, a21_1 = {}, a22_0 = {}, a22_1 = {};
    #pragma unroll
    for (int q = 0; q < 4; ++q) {
        const int boff = ((q * 2 + half) * 32 + r) * 16;     // n = r (+16 per N tile)
        v16h f21a = *(const v16h*)&sWh21[boff];
        v16h f21b = *(const v16h*)&sWh21[boff + 256];
        v16h f22a = *(const v16h*)&sWh22[boff];
        v16h f22b = *(const v16h*)&sWh22[boff + 256];
        a21_0 = __builtin_amdgcn_wmma_f32_16x16x32_f16(false, afr[q], false, f21a, (short)0, a21_0, false, false);
        a21_1 = __builtin_amdgcn_wmma_f32_16x16x32_f16(false, afr[q], false, f21b, (short)0, a21_1, false, false);
        a22_0 = __builtin_amdgcn_wmma_f32_16x16x32_f16(false, afr[q], false, f22a, (short)0, a22_0, false, false);
        a22_1 = __builtin_amdgcn_wmma_f32_16x16x32_f16(false, afr[q], false, f22b, (short)0, a22_1, false, false);
    }

    // ---- bias + relu in C-layout, stage to padded LDS scratch
    const float bb21_0 = b21[r], bb21_1 = b21[r + 16];
    const float bb22_0 = b22[r], bb22_1 = b22[r + 16];
    #pragma unroll
    for (int v = 0; v < 8; ++v) {
        const int rl = v + 8 * half;                 // C/D layout: M = v + 8*(lane>=16), N = lane%16
        sS21[w][rl][r]      = fmaxf(a21_0[v] + bb21_0, 0.0f);
        sS21[w][rl][r + 16] = fmaxf(a21_1[v] + bb21_1, 0.0f);
        sS22[w][rl][r]      = fmaxf(a22_0[v] + bb22_0, 0.0f);
        sS22[w][rl][r + 16] = fmaxf(a22_1[v] + bb22_1, 0.0f);
    }
    __syncthreads();

    // ---- tail: one thread per batch row (heads + sigmoid + CBF clip), exact f32
    if (tid < ROWS_PER_BLOCK && row0 + tid < nB) {
        const int wv = tid >> 4, r2 = tid & 15;
        float s0 = b31[0], s1 = b31[1], z0 = b32[0], z1 = b32[1];
        #pragma unroll
        for (int n = 0; n < 32; ++n) {
            float a  = sS21[wv][r2][n];
            float bq = sS22[wv][r2][n];
            s0 = fmaf(a,  W31[n],      s0);
            s1 = fmaf(a,  W31[32 + n], s1);
            z0 = fmaf(bq, W32[n],      z0);
            z1 = fmaf(bq, W32[32 + n], z1);
        }
        const float x32_0 = 4.0f / (1.0f + expf(-z0));
        const float x32_1 = 4.0f / (1.0f + expf(-z1));
        const float hv = x32_0 * x32_1;

        // xo = (x*std+mean)*std+mean  (reference applies the affine twice)
        float xo[5];
        #pragma unroll
        for (int f = 0; f < 5; ++f) {
            float t0 = fmaf(sX[wv][r2][f], stdv[f], mean[f]);
            xo[f] = fmaf(t0, stdv[f], mean[f]);
        }
        const float px = xo[0], py = xo[1], th = xo[2], oppx = xo[3], oppy = xo[4];
        const float ct = cosf(th), st = sinf(th);

        float upper =  __builtin_inff();
        float lower = -__builtin_inff();
        #pragma unroll
        for (int i = 0; i < 9; ++i) {
            float ox, oy, orr;
            if (i < 8) { ox = obstacles[i*3+0]; oy = obstacles[i*3+1]; orr = obstacles[i*3+2]; }
            else       { ox = oppx;             oy = oppy;             orr = 0.3f; }
            const float dx = px - ox, dy = py - oy;
            const float bar = dx*dx + dy*dy - orr*orr;
            const float g1  = -2.0f * (dx*ct + dy*st);
            const float sg  = (g1 != 0.0f) ? g1 : 1.0f;
            const float ratio = hv * bar / sg;
            upper = (g1 > 0.0f) ? fminf(upper, ratio) : upper;
            lower = (g1 < 0.0f) ? fmaxf(lower, ratio) : lower;
        }
        const float u1 = fminf(fmaxf(-s0, lower), upper);
        const float u2 = -s1;
        const long rg = row0 + tid;
        out[rg * 2 + 0] = u1;
        out[rg * 2 + 1] = u2;
    }
}

extern "C" void kernel_launch(void* const* d_in, const int* in_sizes, int n_in,
                              void* d_out, int out_size, void* d_ws, size_t ws_size,
                              hipStream_t stream) {
    const float* x    = (const float*)d_in[0];
    const float* mean = (const float*)d_in[1];
    const float* stdv = (const float*)d_in[2];
    const float* W1   = (const float*)d_in[3];
    const float* b1   = (const float*)d_in[4];
    const float* W21  = (const float*)d_in[5];
    const float* b21  = (const float*)d_in[6];
    const float* W22  = (const float*)d_in[7];
    const float* b22  = (const float*)d_in[8];
    const float* W31  = (const float*)d_in[9];
    const float* b31  = (const float*)d_in[10];
    const float* W32  = (const float*)d_in[11];
    const float* b32  = (const float*)d_in[12];
    const float* obst = (const float*)d_in[13];
    float* out = (float*)d_out;

    const int nB = in_sizes[0] / 5;
    const int grid = (nB + ROWS_PER_BLOCK - 1) / ROWS_PER_BLOCK;
    barriernet_wmma_kernel<<<grid, THREADS, 0, stream>>>(
        x, mean, stdv, W1, b1, W21, b21, W22, b22, W31, b31, W32, b32, obst, out, nB);
}
